// RoPEMamba_83708912599107
// MI455X (gfx1250) — matrix operand order repaired and checked by reading, not compiled, not run on hardware
//
#include <hip/hip_runtime.h>
#include <hip/hip_bf16.h>
#include <stdint.h>

// ---------------------------------------------------------------------------
// RoPE-Mamba 2D forward for MI455X (gfx1250, wave32, WMMA).
//  - All matmul work (in-proj GEMM, 4x conv-as-implicit-GEMM, out-proj GEMM)
//    through v_wmma_f32_16x16x32_bf16, f32 accumulation.
//  - 64x64 block tiles, 8 wave32 waves, double-buffered LDS, one barrier per
//    K-step. A-tile staged via GLOBAL_LOAD_ASYNC_TO_LDS_B128 when available.
//  - Ragged N (2088) padded to 2112 so the hot loops have no edge handling.
// ---------------------------------------------------------------------------

#define D_MODEL   256
#define NHEADS    8
#define HEADDIM   32
#define HH        64
#define WWI       64
#define LSEQ      4096          // 64*64
#define BATCH     2
#define ROWS      (BATCH*LSEQ)  // 8192
#define PER       522           // 256 + 258 + 8
#define NPROJ     2088          // 4*PER
#define NPAD      2112          // NPROJ padded to multiple of 64
#define CCONV     512           // 2*D_MODEL

typedef __attribute__((ext_vector_type(8)))  __bf16 v8bf;
typedef __attribute__((ext_vector_type(16))) __bf16 v16bf;
typedef __attribute__((ext_vector_type(8)))  float  v8f;
typedef __attribute__((ext_vector_type(4)))  int    v4i;

union BfU { v8bf v; unsigned short s[8]; };

#if defined(__has_builtin)
#if __has_builtin(__builtin_amdgcn_global_load_async_to_lds_b128) && \
    __has_builtin(__builtin_amdgcn_s_wait_asynccnt)
#define CDNA5_ASYNC 1
#endif
#endif
#ifndef CDNA5_ASYNC
#define CDNA5_ASYNC 0
#endif

#if CDNA5_ASYNC
// Signature per clang-22 diagnostic: (int4 AS1*, int4 AS3*, imm int, imm int)
__device__ __forceinline__ void async_ld_b128(void* lds, const void* g) {
  __builtin_amdgcn_global_load_async_to_lds_b128(
      (__attribute__((address_space(1))) v4i*)(unsigned long long)(uintptr_t)g,
      (__attribute__((address_space(3))) v4i*)(unsigned int)(uintptr_t)lds,
      0, 0);
}
__device__ __forceinline__ void async_wait0() {
  __builtin_amdgcn_s_wait_asynccnt(0);
}
#else
__device__ __forceinline__ void async_wait0() {}
#endif

__device__ __forceinline__ unsigned short f2bf(float f) {
  unsigned int u = __float_as_uint(f);
  unsigned int r = u + 0x7FFFu + ((u >> 16) & 1u);   // round-to-nearest-even
  return (unsigned short)(r >> 16);
}

// ---------------------------------------------------------------------------
// Wave-level WMMA micro-step: one 16x32 A fragment vs two 32x16 B fragments.
// sA: [64 m][32 k] bf16 (row stride 64B), sB: [64 n][32 k] bf16 (transposed).
// Fragment k order: contiguous, lane half h holds k = 16h..16h+15 in both A and B (wave32).
// ---------------------------------------------------------------------------
__device__ __forceinline__ void wmma_step(const unsigned short* sA,
                                          const unsigned short* sB,
                                          int lane, int wm, int wn,
                                          v8f& acc0, v8f& acc1)
{
  const int mrow = wm + (lane & 15);
  const int b2   = (lane >> 4) << 4;      // 0 or 16: A K-half select
  v8bf alo = *(const v8bf*)(sA + mrow * 32 + b2);        // K = b2 .. b2+7
  v8bf ahi = *(const v8bf*)(sA + mrow * 32 + b2 + 8);    // K = b2+8 .. b2+15
  v16bf af = __builtin_shufflevector(alo, ahi,
      0,1,2,3,4,5,6,7,8,9,10,11,12,13,14,15);

  const int kb = (lane >> 4) << 4;        // 0 or 16: B K-half select
  const int n0 = wn + (lane & 15);
  v8bf b0lo = *(const v8bf*)(sB + n0 * 32 + kb);
  v8bf b0hi = *(const v8bf*)(sB + n0 * 32 + kb + 8);
  v16bf bf0 = __builtin_shufflevector(b0lo, b0hi,
      0,1,2,3,4,5,6,7,8,9,10,11,12,13,14,15);
  v8bf b1lo = *(const v8bf*)(sB + (n0 + 16) * 32 + kb);
  v8bf b1hi = *(const v8bf*)(sB + (n0 + 16) * 32 + kb + 8);
  v16bf bf1 = __builtin_shufflevector(b1lo, b1hi,
      0,1,2,3,4,5,6,7,8,9,10,11,12,13,14,15);

  acc0 = __builtin_amdgcn_wmma_f32_16x16x32_bf16(false, af, false, bf0,
                                                 (short)0, acc0, false, false);
  acc1 = __builtin_amdgcn_wmma_f32_16x16x32_bf16(false, af, false, bf1,
                                                 (short)0, acc1, false, false);
}

// ---------------------------------------------------------------------------
// bf16 GEMM: C[M,N](f32) = A[M,K] * B[K,N], row-major. M%64==0, N%64==0,
// K%32==0 (caller guarantees; N is pre-padded). 256 threads, tile 64x64.
// ---------------------------------------------------------------------------
__global__ __launch_bounds__(256) void k_gemm_bf16(
    const unsigned short* __restrict__ A, const unsigned short* __restrict__ B,
    float* __restrict__ C, int K, int lda, int ldb, int ldc)
{
  __shared__ __align__(16) unsigned short sA[2][64 * 32];
  __shared__ __align__(16) unsigned short sB[2][64 * 32];
  const int tid = threadIdx.x, lane = tid & 31, wave = tid >> 5;
  const int tileM = blockIdx.x * 64, tileN = blockIdx.y * 64;
  const int wm = (wave & 3) * 16, wn = (wave >> 2) * 32;
  v8f acc0 = {0.f,0.f,0.f,0.f,0.f,0.f,0.f,0.f};
  v8f acc1 = {0.f,0.f,0.f,0.f,0.f,0.f,0.f,0.f};

  // A staging: thread -> (row ar, 8 contiguous k at ac)
  const int ar = tid >> 2, ac = (tid & 3) * 8;
  const unsigned short* aG = A + (size_t)(tileM + ar) * lda + ac;
  // B staging: thread -> (k pair 2*kp, 4 contiguous n at n0); transpose into LDS
  const int kp = tid & 15, n0 = (tid >> 4) * 4;
  const unsigned short* bG = B + (size_t)(2 * kp) * ldb + tileN + n0;

  const int nk = K / 32;
  ushort4 b0r, b1r;
#if !CDNA5_ASYNC
  v8bf aReg;
#endif

  auto issueStage = [&](int kk, int buf) {
#if CDNA5_ASYNC
    async_ld_b128(&sA[buf][ar * 32 + ac], aG + kk);
#else
    aReg = *(const v8bf*)(aG + kk);
#endif
    __builtin_prefetch(aG + kk + 64, 0, 1);
    b0r = *(const ushort4*)(bG + (size_t)kk * ldb);
    b1r = *(const ushort4*)(bG + (size_t)kk * ldb + ldb);
  };
  auto commitStage = [&](int buf) {
#if !CDNA5_ASYNC
    *(v8bf*)(&sA[buf][ar * 32 + ac]) = aReg;
#endif
    unsigned short* d = &sB[buf][n0 * 32 + 2 * kp];
    *(uint32_t*)(d +  0) = (uint32_t)b0r.x | ((uint32_t)b1r.x << 16);
    *(uint32_t*)(d + 32) = (uint32_t)b0r.y | ((uint32_t)b1r.y << 16);
    *(uint32_t*)(d + 64) = (uint32_t)b0r.z | ((uint32_t)b1r.z << 16);
    *(uint32_t*)(d + 96) = (uint32_t)b0r.w | ((uint32_t)b1r.w << 16);
  };

  issueStage(0, 0);
  commitStage(0);
  async_wait0();
  __syncthreads();
  for (int k = 0; k < nk; ++k) {
    const int cur = k & 1;
    if (k + 1 < nk) issueStage((k + 1) * 32, cur ^ 1);  // buf cur^1 free since bar(k-1)
    wmma_step(sA[cur], sB[cur], lane, wm, wn, acc0, acc1);
    if (k + 1 < nk) { commitStage(cur ^ 1); async_wait0(); }
    __syncthreads();
  }

  const int mrow = tileM + wm + ((lane >> 4) << 3);
  const int nc = tileN + wn + (lane & 15);
  for (int v = 0; v < 8; ++v) {
    C[(size_t)(mrow + v) * ldc + nc]      = acc0[v];
    C[(size_t)(mrow + v) * ldc + nc + 16] = acc1[v];
  }
}

// ---------------------------------------------------------------------------
// 3x3 SAME conv as implicit GEMM. In: NHWC bf16 (ROWS x 512). Wc: packed
// [dir][ky][kx][i][o] bf16. Out: ROWS x 512 f32. K = 9*512, step 32.
// ---------------------------------------------------------------------------
__global__ __launch_bounds__(256) void k_conv_wmma(
    const unsigned short* __restrict__ In, const unsigned short* __restrict__ Wc,
    float* __restrict__ Out, int dir)
{
  __shared__ __align__(16) unsigned short sA[2][64 * 32];
  __shared__ __align__(16) unsigned short sB[2][64 * 32];
  const int tid = threadIdx.x, lane = tid & 31, wave = tid >> 5;
  const int tileM = blockIdx.x * 64, tileN = blockIdx.y * 64;
  const int wm = (wave & 3) * 16, wn = (wave >> 2) * 32;
  v8f acc0 = {0.f,0.f,0.f,0.f,0.f,0.f,0.f,0.f};
  v8f acc1 = {0.f,0.f,0.f,0.f,0.f,0.f,0.f,0.f};
  const int ar = tid >> 2, ac = (tid & 3) * 8;
  const int kp = tid & 15, n0 = (tid >> 4) * 4;
  const int px = tileM + ar;
  const int pb = px >> 12, prem = px & 4095, py = prem >> 6, pxx = prem & 63;
  const unsigned short* wDir = Wc + (size_t)(dir * 9) * CCONV * CCONV + tileN + n0;

  const int nk = 9 * (CCONV / 32);          // 144 K-steps
  ushort4 b0r, b1r;
  bool stInb = false;
#if !CDNA5_ASYNC
  v8bf aReg;
#endif

  auto issueStage = [&](int s, int buf) {
    const int kpos = s >> 4;                // 0..8
    const int kc = (s & 15) << 5;           // 0..480
    const int ky = kpos / 3, kx = kpos - ky * 3;
    const int yy = py + ky - 1, xx = pxx + kx - 1;
    stInb = (yy >= 0) & (yy < HH) & (xx >= 0) & (xx < WWI);
    if (stInb) {
      const unsigned short* gp =
          In + ((size_t)(pb * LSEQ + yy * WWI + xx)) * CCONV + kc + ac;
#if CDNA5_ASYNC
      async_ld_b128(&sA[buf][ar * 32 + ac], gp);
#else
      aReg = *(const v8bf*)gp;
#endif
    }
    const unsigned short* wb =
        wDir + (size_t)(kpos * CCONV + kc + 2 * kp) * CCONV;
    b0r = *(const ushort4*)wb;
    b1r = *(const ushort4*)(wb + CCONV);
  };
  auto commitStage = [&](int buf) {
#if CDNA5_ASYNC
    if (!stInb) {
      BfU z; for (int j = 0; j < 8; ++j) z.s[j] = 0;
      *(v8bf*)(&sA[buf][ar * 32 + ac]) = z.v;
    }
#else
    if (!stInb) { BfU z; for (int j = 0; j < 8; ++j) z.s[j] = 0; aReg = z.v; }
    *(v8bf*)(&sA[buf][ar * 32 + ac]) = aReg;
#endif
    unsigned short* d = &sB[buf][n0 * 32 + 2 * kp];
    *(uint32_t*)(d +  0) = (uint32_t)b0r.x | ((uint32_t)b1r.x << 16);
    *(uint32_t*)(d + 32) = (uint32_t)b0r.y | ((uint32_t)b1r.y << 16);
    *(uint32_t*)(d + 64) = (uint32_t)b0r.z | ((uint32_t)b1r.z << 16);
    *(uint32_t*)(d + 96) = (uint32_t)b0r.w | ((uint32_t)b1r.w << 16);
  };

  issueStage(0, 0);
  commitStage(0);
  async_wait0();
  __syncthreads();
  for (int s = 0; s < nk; ++s) {
    const int cur = s & 1;
    if (s + 1 < nk) issueStage(s + 1, cur ^ 1);
    wmma_step(sA[cur], sB[cur], lane, wm, wn, acc0, acc1);
    if (s + 1 < nk) { commitStage(cur ^ 1); async_wait0(); }
    __syncthreads();
  }

  const int mrow = tileM + wm + ((lane >> 4) << 3);
  const int nc = tileN + wn + (lane & 15);
  for (int v = 0; v < 8; ++v) {
    Out[(size_t)(mrow + v) * CCONV + nc]      = acc0[v];
    Out[(size_t)(mrow + v) * CCONV + nc + 16] = acc1[v];
  }
}

// ---------------------------------------------------------------------------
// Elementwise / pack kernels
// ---------------------------------------------------------------------------
__global__ void k_pack_x(const float* __restrict__ x,
                         unsigned short* __restrict__ Xbf, int n)
{
  int i = blockIdx.x * blockDim.x + threadIdx.x;
  if (i >= n) return;
  int c = i & (D_MODEL - 1);
  int r = i >> 8;                       // b*L + l
  int b = r >> 12, l = r & 4095;
  Xbf[i] = f2bf(x[(((size_t)b * D_MODEL + c) << 12) + l]);
}

// W_in (256 x 2088) -> bf16 padded to (256 x 2112) with zero columns
__global__ void k_pack_win(const float* __restrict__ src,
                           unsigned short* __restrict__ dst, int n)
{
  int i = blockIdx.x * blockDim.x + threadIdx.x;
  if (i >= n) return;
  int r = i / NPAD, c = i - r * NPAD;
  dst[i] = (c < NPROJ) ? f2bf(src[(size_t)r * NPROJ + c]) : (unsigned short)0;
}

__global__ void k_pack_f32bf(const float* __restrict__ src,
                             unsigned short* __restrict__ dst, int n)
{
  int i = blockIdx.x * blockDim.x + threadIdx.x;
  if (i < n) dst[i] = f2bf(src[i]);
}

// conv_w (4, O=512, I=512, 3, 3) -> Wc [d][ky][kx][i][o]
__global__ void k_pack_convw(const float* __restrict__ cw,
                             unsigned short* __restrict__ Wc, int n)
{
  int i = blockIdx.x * blockDim.x + threadIdx.x;
  if (i >= n) return;
  int o  = i & 511;
  int ii = (i >> 9) & 511;
  int kk = (i >> 18) % 9;
  int d  = i / (512 * 512 * 9);
  int ky = kk / 3, kx = kk % 3;
  Wc[i] = f2bf(cw[((((size_t)d * 512 + o) * 512 + ii) * 3 + ky) * 3 + kx]);
}

// zxbcdt row (padded ld NPAD) -> conv input NHWC bf16 for one direction:
// ch 0..255 = xd (blk[256:512]), ch 256..511 = z (blk[0:256])
__global__ void k_pack_convin(const float* __restrict__ zx,
                              unsigned short* __restrict__ cin, int dir, int n)
{
  int i = blockIdx.x * blockDim.x + threadIdx.x;
  if (i >= n) return;
  int c = i & 511;
  int r = i >> 9;
  int src = (c < 256) ? (256 + c) : (c - 256);
  cin[i] = f2bf(zx[(size_t)r * NPAD + dir * PER + src]);
}

__global__ void k_gelu_bn(float* __restrict__ v,
                          const float* __restrict__ g, const float* __restrict__ b,
                          const float* __restrict__ m, const float* __restrict__ vr,
                          int dir, int n)
{
  int i = blockIdx.x * blockDim.x + threadIdx.x;
  if (i >= n) return;
  int c = (i & 511) + dir * CCONV;
  float x = v[i];
  float u = 0.7978845608028654f * (x + 0.044715f * x * x * x);
  x = 0.5f * x * (1.0f + tanhf(u));            // tanh-approx GELU (jax default)
  x = (x - m[c]) * rsqrtf(vr[c] + 1e-3f) * g[c] + b[c];
  v[i] = x;
}

// ---------------------------------------------------------------------------
// Fused: morton reorder (+flip) gather, RoPE, softplus, ds=1 SSM recurrence,
// C-readout + D-skip, SiLU(z) gating, inverse reorder scatter, bf16 repack.
// One wave per (batch, head); lanes = headdim.
// ---------------------------------------------------------------------------
__global__ void k_scan(const float* __restrict__ zx,
                       const float* __restrict__ conv,
                       const int* __restrict__ mH, const int* __restrict__ mV,
                       const float* __restrict__ fcos, const float* __restrict__ fsin,
                       const float* __restrict__ dt_bias,
                       const float* __restrict__ A_log,
                       const float* __restrict__ Dp,
                       unsigned short* __restrict__ A2, int dir)
{
  const int lane = threadIdx.x;             // 0..31 (= p)
  const int bh = blockIdx.x;                // 0..15
  const int b = bh >> 3, h = bh & 7;
  const int* idx = (dir < 2) ? mH : mV;
  const bool flip = dir & 1;
  const float Acoef = -__expf(A_log[dir * NHEADS + h]);
  const float dbias = dt_bias[dir * NHEADS + h];
  const float Dd = Dp[dir * NHEADS + h];
  const int j = lane >> 1;
  float hstate = 0.f;

  for (int l = 0; l < LSEQ; ++l) {
    const int t = idx[flip ? (LSEQ - 1 - l) : l];
    const size_t rz = (size_t)(b * LSEQ + t) * NPAD + (size_t)dir * PER;
    const float dtr = zx[rz + 514 + h];
    const float Bd  = zx[rz + 512];
    const float Cd  = zx[rz + 513];
    const float s0 = dtr + dbias;
    const float dtv = (s0 > 20.f) ? s0 : log1pf(__expf(s0));
    const float a = __expf(dtv * Acoef);

    const size_t rc = (size_t)(b * LSEQ + t) * CCONV;
    const float xv = conv[rc + h * HEADDIM + lane];
    const float zv = conv[rc + 256 + h * HEADDIM + lane];
    const float xo = __shfl_xor(xv, 1, 32);
    const float cc = fcos[l * 16 + j], ss = fsin[l * 16 + j];
    const float xr = (lane & 1) ? (xo * ss + xv * cc)     // imag
                                : (xv * cc - xo * ss);    // real

    hstate = a * hstate + dtv * xr * Bd;
    float y = hstate * Cd + Dd * xr;
    const float sig = 1.f / (1.f + __expf(-zv));
    y *= zv * sig;
    A2[(size_t)(b * LSEQ + t) * (4 * D_MODEL) + dir * D_MODEL + h * HEADDIM + lane]
        = f2bf(y);
  }
}

// ---------------------------------------------------------------------------
// LayerNorm over 256 + transpose to (B, C, H, W). One block per token row.
// ---------------------------------------------------------------------------
__global__ __launch_bounds__(256) void k_ln_out(
    const float* __restrict__ o2, const float* __restrict__ g,
    const float* __restrict__ bt, float* __restrict__ out)
{
  __shared__ float s1[8], s2[8];
  const int r = blockIdx.x, tid = threadIdx.x;
  const float v = o2[(size_t)r * D_MODEL + tid];
  float a = v, q = v * v;
  for (int off = 16; off; off >>= 1) {
    a += __shfl_down(a, off, 32);
    q += __shfl_down(q, off, 32);
  }
  if ((tid & 31) == 0) { s1[tid >> 5] = a; s2[tid >> 5] = q; }
  __syncthreads();
  if (tid == 0) {
    float ta = 0.f, tq = 0.f;
    for (int i = 0; i < 8; ++i) { ta += s1[i]; tq += s2[i]; }
    const float mu = ta * (1.f / D_MODEL);
    s1[0] = mu;
    s2[0] = tq * (1.f / D_MODEL) - mu * mu;
  }
  __syncthreads();
  const float mu = s1[0], var = s2[0];
  const int b = r >> 12, l = r & 4095;
  out[(((size_t)b * D_MODEL + tid) << 12) + l] =
      (v - mu) * rsqrtf(var + 1e-6f) * g[tid] + bt[tid];
}

// ---------------------------------------------------------------------------
extern "C" void kernel_launch(void* const* d_in, const int* in_sizes, int n_in,
                              void* d_out, int out_size, void* d_ws, size_t ws_size,
                              hipStream_t stream) {
  (void)in_sizes; (void)n_in; (void)out_size; (void)ws_size;
  const float* x      = (const float*)d_in[0];
  const float* fcos   = (const float*)d_in[1];
  const float* fsin   = (const float*)d_in[2];
  const int*   mH     = (const int*)d_in[3];
  const int*   mV     = (const int*)d_in[4];
  // d_in[5], d_in[6] (inv maps) unused: inverse reorder fused into k_scan.
  const float* W_in   = (const float*)d_in[7];
  const float* conv_w = (const float*)d_in[8];
  const float* bn_g   = (const float*)d_in[9];
  const float* bn_b   = (const float*)d_in[10];
  const float* bn_m   = (const float*)d_in[11];
  const float* bn_v   = (const float*)d_in[12];
  const float* dtb    = (const float*)d_in[13];
  const float* A_log  = (const float*)d_in[14];
  const float* Dp     = (const float*)d_in[15];
  const float* W_out  = (const float*)d_in[16];
  const float* ln_g   = (const float*)d_in[17];
  const float* ln_b   = (const float*)d_in[18];
  float* out = (float*)d_out;

  char* ws = (char*)d_ws;
  size_t off = 0;
  auto take = [&](size_t bytes) -> char* {
    char* p = ws + off;
    off = (off + bytes + 255) & ~(size_t)255;
    return p;
  };
  unsigned short* Xbf   = (unsigned short*)take((size_t)ROWS * D_MODEL * 2);
  unsigned short* Winb  = (unsigned short*)take((size_t)D_MODEL * NPAD * 2);
  unsigned short* Woutb = (unsigned short*)take((size_t)4 * D_MODEL * D_MODEL * 2);
  float*          ZX    = (float*)take((size_t)ROWS * NPAD * 4);
  unsigned short* Wcb   = (unsigned short*)take((size_t)4 * 9 * CCONV * CCONV * 2);
  unsigned short* CIn   = (unsigned short*)take((size_t)ROWS * CCONV * 2);
  float*          COut  = (float*)take((size_t)ROWS * CCONV * 4);
  unsigned short* A2    = (unsigned short*)take((size_t)ROWS * 4 * D_MODEL * 2);
  float*          O2    = (float*)take((size_t)ROWS * D_MODEL * 4);

  const int T = 256;
  // pack operands to bf16
  k_pack_x<<<(ROWS * D_MODEL) / T, T, 0, stream>>>(x, Xbf, ROWS * D_MODEL);
  k_pack_win<<<(D_MODEL * NPAD) / T, T, 0, stream>>>(W_in, Winb, D_MODEL * NPAD);
  k_pack_f32bf<<<(4 * D_MODEL * D_MODEL + T - 1) / T, T, 0, stream>>>(W_out, Woutb, 4 * D_MODEL * D_MODEL);
  k_pack_convw<<<(4 * 9 * CCONV * CCONV) / T, T, 0, stream>>>(conv_w, Wcb, 4 * 9 * CCONV * CCONV);

  // in-projection: (8192 x 256) x (256 x 2112pad)
  k_gemm_bf16<<<dim3(ROWS / 64, NPAD / 64), T, 0, stream>>>(
      Xbf, Winb, ZX, D_MODEL, D_MODEL, NPAD, NPAD);

  for (int d = 0; d < 4; ++d) {
    k_pack_convin<<<(ROWS * CCONV) / T, T, 0, stream>>>(ZX, CIn, d, ROWS * CCONV);
    k_conv_wmma<<<dim3(ROWS / 64, CCONV / 64), T, 0, stream>>>(CIn, Wcb, COut, d);
    k_gelu_bn<<<(ROWS * CCONV) / T, T, 0, stream>>>(COut, bn_g, bn_b, bn_m, bn_v, d, ROWS * CCONV);
    k_scan<<<BATCH * NHEADS, 32, 0, stream>>>(ZX, COut, mH, mV, fcos, fsin,
                                              dtb, A_log, Dp, A2, d);
  }

  // out-projection: (8192 x 1024) x (1024 x 256)
  k_gemm_bf16<<<dim3(ROWS / 64, D_MODEL / 64), T, 0, stream>>>(
      A2, Woutb, O2, 4 * D_MODEL, 4 * D_MODEL, D_MODEL, D_MODEL);

  k_ln_out<<<ROWS, T, 0, stream>>>(O2, ln_g, ln_b, out);
}
